// BaseWindowAttention_51977694216822
// MI455X (gfx1250) — matrix operand order repaired and checked by reading, not compiled, not run on hardware
//
#include <hip/hip_runtime.h>
#include <stdint.h>

// ---------------- problem constants ----------------
#define WSZ   8
#define WSQ   64                  // pixels per window
#define HEADS 8
#define HD    32
#define DIMC  256
#define NQKV  768
#define BLD   8                   // b*l
#define HPIX  128
#define NWIN  256                 // 16*16 windows per (b,l)
#define ROWS  (BLD * HPIX * HPIX) // 131072 pixel rows
// one of q/k/v in bf16 windowed layout, SEL_STRIDE elements each
#define SEL_STRIDE ((size_t)BLD * HEADS * NWIN * WSQ * HD) // 33,554,432 elems

// ---------------- WMMA types (CDNA5, wave32) ----------------
typedef __attribute__((ext_vector_type(16))) __bf16 bf16x16;
typedef __attribute__((ext_vector_type(2)))  __bf16 bf16x2;
typedef __attribute__((ext_vector_type(8)))  float  f32x8;

union FragAB { bf16x16 v; unsigned int u[8]; };
union FragC  { f32x8   v; float        f[8]; };

// ---- native bf16 conversion (v_cvt_bf16 / v_cvt_pk_bf16_f32) ----
__device__ __forceinline__ unsigned short f2bf(float x) {
  union { __bf16 b; unsigned short s; } t;
  t.b = (__bf16)x;
  return t.s;
}
__device__ __forceinline__ unsigned int pack2bf(float a, float b) {
#if __has_builtin(__builtin_amdgcn_cvt_pk_bf16_f32)
  union { bf16x2 v; unsigned int u; } t;
  t.v = __builtin_amdgcn_cvt_pk_bf16_f32(a, b);
  return t.u;
#else
  union { __bf16 b[2]; unsigned int u; } t;
  t.b[0] = (__bf16)a; t.b[1] = (__bf16)b;
  return t.u;
#endif
}

// ---- gfx1250 async global->LDS copy (ASYNCcnt), guarded with sync fallback ----
// Builtin prototype (from hipcc diagnostic): takes
//   int __attribute__((vector_size(16))) addrspace(1)* (global source)
//   and the LDS destination; match the GCC-style vector pointee exactly.
typedef int v4i_vs __attribute__((vector_size(16)));
typedef __attribute__((address_space(1))) v4i_vs* gp128_t;
typedef __attribute__((address_space(3))) v4i_vs* lp128_t;
#if __has_builtin(__builtin_amdgcn_global_load_async_to_lds_b128)
#define HAS_ASYNC_LDS 1
#else
#define HAS_ASYNC_LDS 0
#endif

__device__ __forceinline__ void async_copy16(const unsigned short* g, unsigned short* l) {
#if HAS_ASYNC_LDS
  __builtin_amdgcn_global_load_async_to_lds_b128((gp128_t)g, (lp128_t)l, 0, 0);
#else
  *(uint4*)l = *(const uint4*)g;
#endif
}
__device__ __forceinline__ void wait_async0() {
#if HAS_ASYNC_LDS
#if __has_builtin(__builtin_amdgcn_s_wait_asynccnt)
  __builtin_amdgcn_s_wait_asynccnt(0);
#else
  asm volatile("s_wait_asynccnt 0x0" ::: "memory");
#endif
#endif
}

// A fragment: 16(M) x 32(K) bf16, row-major source, rowStride in elements.
// ISA 7.12.2: lanes 0-15 hold K 0-7 (v0-3) & 16-23 (v4-7); lanes 16-31 K 8-15 & 24-31.
__device__ __forceinline__ void load_frag_a(const unsigned short* base, int rowStride,
                                            int l16, int half, FragAB& f) {
  const unsigned short* r = base + l16 * rowStride;
  uint4 lo = *(const uint4*)(r + half * 8);
  uint4 hi = *(const uint4*)(r + 16 + half * 8);
  f.u[0] = lo.x; f.u[1] = lo.y; f.u[2] = lo.z; f.u[3] = lo.w;
  f.u[4] = hi.x; f.u[5] = hi.y; f.u[6] = hi.z; f.u[7] = hi.w;
}
// B fragment: 32(K) x 16(N) bf16 sourced from an N-major array (row n = one B column,
// K contiguous). lanes 0-15: K 0-7 & 16-23; lanes 16-31: K 8-15 & 24-31; lane%16 = N column.
__device__ __forceinline__ void load_frag_b(const unsigned short* base, int rowStride,
                                            int l16, int half, FragAB& f) {
  const unsigned short* r = base + l16 * rowStride;
  uint4 lo = *(const uint4*)(r + half * 8);
  uint4 hi = *(const uint4*)(r + 16 + half * 8);
  f.u[0] = lo.x; f.u[1] = lo.y; f.u[2] = lo.z; f.u[3] = lo.w;
  f.u[4] = hi.x; f.u[5] = hi.y; f.u[6] = hi.z; f.u[7] = hi.w;
}
__device__ __forceinline__ f32x8 wmma_bf16(const FragAB& a, const FragAB& b, f32x8 c) {
  return __builtin_amdgcn_wmma_f32_16x16x32_bf16(false, a.v, false, b.v,
                                                 (short)0, c, false, false);
}

// ================= kernel 1: qkv = x @ W_qkv (f32 in, bf16 windowed out) =================
// grid (NQKV/64, ROWS/64), block 256 (8 waves, 4x2 wave grid, 16x32 C per wave)
// q,k stored [bl][head][win][pos][d]; v stored TRANSPOSED [bl][head][win][d][pos]
// so the attention kernel needs no in-kernel transpose.
__global__ __launch_bounds__(256) void qkv_proj_kernel(
    const float* __restrict__ x, const float* __restrict__ Wqkv,
    unsigned short* __restrict__ qkv_ws) {
  __shared__ unsigned short As[64 * 40];   // bf16 A tile, row stride 40 (pad)
  __shared__ unsigned short Bs[64 * 40];   // bf16 B tile, N-major [n][k]
  const int tid  = threadIdx.x;
  const int lane = tid & 31, l16 = lane & 15, half = lane >> 4;
  const int wid  = tid >> 5, wy = wid >> 1, wx = wid & 1;
  const int rowBase = blockIdx.y * 64;
  const int colBase = blockIdx.x * 64;

  FragC acc[2];
  #pragma unroll
  for (int t = 0; t < 2; ++t)
    #pragma unroll
    for (int i = 0; i < 8; ++i) acc[t].f[i] = 0.f;

  for (int k0 = 0; k0 < DIMC; k0 += 32) {
    #pragma unroll
    for (int rep = 0; rep < 2; ++rep) {
      int fidx = tid + rep * 256;             // float4 index in [0,512)
      // ---- A: 64x32 f32 -> bf16 LDS ----
      int r = fidx >> 3, c4 = fidx & 7;
      const float* ap = x + (size_t)(rowBase + r) * DIMC + k0 + c4 * 4;
      float4 vx = *(const float4*)ap;
      if (k0 + 32 < DIMC) __builtin_prefetch(ap + 32, 0, 1);  // global_prefetch_b8
      *(uint2*)(&As[r * 40 + c4 * 4]) = make_uint2(pack2bf(vx.x, vx.y), pack2bf(vx.z, vx.w));
      // ---- B: 32x64 f32 -> bf16 LDS transposed (N-major) ----
      int k = fidx >> 4, n4 = fidx & 15;
      float4 vw = *(const float4*)(Wqkv + (size_t)(k0 + k) * NQKV + colBase + n4 * 4);
      Bs[(n4 * 4 + 0) * 40 + k] = f2bf(vw.x);
      Bs[(n4 * 4 + 1) * 40 + k] = f2bf(vw.y);
      Bs[(n4 * 4 + 2) * 40 + k] = f2bf(vw.z);
      Bs[(n4 * 4 + 3) * 40 + k] = f2bf(vw.w);
    }
    __syncthreads();
    FragAB a, b0, b1;
    load_frag_a(As + wy * 16 * 40, 40, l16, half, a);
    load_frag_b(Bs + (wx * 32) * 40, 40, l16, half, b0);
    load_frag_b(Bs + (wx * 32 + 16) * 40, 40, l16, half, b1);
    acc[0].v = wmma_bf16(a, b0, acc[0].v);
    acc[1].v = wmma_bf16(a, b1, acc[1].v);
    __syncthreads();
  }

  // epilogue: scatter bf16 into windowed q/k/v
  #pragma unroll
  for (int t = 0; t < 2; ++t) {
    int cg   = colBase + wx * 32 + t * 16 + l16;
    int sel  = cg >> 8, rem = cg & 255, head = rem >> 5, d = rem & 31;
    unsigned short* dst = qkv_ws + (size_t)sel * SEL_STRIDE;
    #pragma unroll
    for (int i = 0; i < 8; ++i) {
      int rg  = rowBase + wy * 16 + i + half * 8;
      int bl  = rg >> 14, pix = rg & 16383, h = pix >> 7, w = pix & 127;
      int win = ((h >> 3) << 4) | (w >> 3);
      int pos = ((h & 7) << 3) | (w & 7);
      size_t base = ((size_t)(bl * HEADS + head)) * NWIN + win;
      size_t off  = (sel == 2) ? (base * HD + d) * WSQ + pos     // v: [d][pos]
                               : (base * WSQ + pos) * HD + d;    // q,k: [pos][d]
      dst[off] = f2bf(acc[t].f[i]);
    }
  }
}

// ================= kernel 2: per-window attention =================
// grid (NWIN, HEADS, BLD), block 128 (4 waves; wave w owns query rows w*16..w*16+15)
__global__ __launch_bounds__(128) void window_attn_kernel(
    const unsigned short* __restrict__ qkv_ws, const float* __restrict__ pos_emb,
    unsigned short* __restrict__ attn_ws) {
  __shared__ unsigned short kS[64 * 40];    // k tile [pos][d], padded
  __shared__ unsigned short vS[32 * 72];    // v tile already transposed [d][pos], padded
  __shared__ unsigned short Pb[64 * 72];    // softmax probs bf16
  __shared__ float posb[15 * 15];
  const int win = blockIdx.x, head = blockIdx.y, bl = blockIdx.z;
  const int tid  = threadIdx.x;
  const int lane = tid & 31, l16 = lane & 15, half = lane >> 4;
  const int wv   = tid >> 5;                // wave id = query row-tile

  const size_t tile = (((size_t)(bl * HEADS + head)) * NWIN + win) * (WSQ * HD);
  const unsigned short* qb = qkv_ws + tile;
  const unsigned short* kb = qkv_ws + SEL_STRIDE + tile;
  const unsigned short* vb = qkv_ws + 2 * SEL_STRIDE + tile;   // [d][pos]

  for (int p = tid; p < 225; p += 128) posb[p] = pos_emb[p];
  // async-stage k and v tiles into padded LDS (ASYNCcnt path on gfx1250)
  #pragma unroll
  for (int rep = 0; rep < 2; ++rep) {
    int ch = tid + rep * 128;               // 16B chunk id in [0,256)
    async_copy16(kb + ch * 8, &kS[(ch >> 2) * 40 + (ch & 3) * 8]);
    async_copy16(vb + ch * 8, &vS[(ch >> 3) * 72 + (ch & 7) * 8]);
  }
  wait_async0();
  __syncthreads();

  // S(16x64) = q_tile(16x32) @ k^T : 4 WMMAs (B columns of k^T = rows of k)
  FragAB qf;
  load_frag_a(qb + wv * 16 * HD, HD, l16, half, qf);   // q direct from global (contiguous)
  FragC s[4];
  #pragma unroll
  for (int j = 0; j < 4; ++j) {
    FragAB kf;
    load_frag_b(kS + (j * 16) * 40, 40, l16, half, kf);
    f32x8 z = {0, 0, 0, 0, 0, 0, 0, 0};
    s[j].v = wmma_bf16(qf, kf, z);
  }

  // fused rel-pos bias + row softmax. C layout: lane holds col j*16+l16,
  // rows wv*16+i (lanes 0-15) / +8 (lanes 16-31); butterflies over masks 1..8
  // stay inside each 16-lane half (wave32).
  const float scale = 0.17677669529663687f;   // 32^-0.5
  #pragma unroll
  for (int i = 0; i < 8; ++i) {
    int pi = wv * 16 + i + half * 8;
    int xi = pi >> 3, yi = pi & 7;
    float m = -3.0e38f;
    #pragma unroll
    for (int j = 0; j < 4; ++j) {
      int pj = j * 16 + l16;
      int xj = pj >> 3, yj = pj & 7;
      float v = s[j].f[i] * scale + posb[(xj - xi + 7) * 15 + (yj - yi + 7)];
      s[j].f[i] = v;
      m = fmaxf(m, v);
    }
    for (int off = 1; off < 16; off <<= 1) m = fmaxf(m, __shfl_xor(m, off, 32));
    float sum = 0.f;
    #pragma unroll
    for (int j = 0; j < 4; ++j) { float e = __expf(s[j].f[i] - m); s[j].f[i] = e; sum += e; }
    for (int off = 1; off < 16; off <<= 1) sum += __shfl_xor(sum, off, 32);
    float inv = 1.0f / sum;
    #pragma unroll
    for (int j = 0; j < 4; ++j) s[j].f[i] *= inv;
  }

  // re-fragment P through LDS as bf16 (C layout -> A layout transpose)
  #pragma unroll
  for (int j = 0; j < 4; ++j)
    #pragma unroll
    for (int i = 0; i < 8; ++i) {
      int row = wv * 16 + i + half * 8;
      Pb[row * 72 + j * 16 + l16] = f2bf(s[j].f[i]);
    }
  __syncthreads();

  // O(16x32) = P(16x64) @ V(64x32): K split in 2, N split in 2 -> 4 WMMAs
  FragC o[2];
  #pragma unroll
  for (int t = 0; t < 2; ++t)
    #pragma unroll
    for (int i = 0; i < 8; ++i) o[t].f[i] = 0.f;
  #pragma unroll
  for (int ka = 0; ka < 2; ++ka) {
    FragAB pf;
    load_frag_a(Pb + wv * 16 * 72 + ka * 32, 72, l16, half, pf);
    #pragma unroll
    for (int t = 0; t < 2; ++t) {
      FragAB vf;
      load_frag_b(vS + (t * 16) * 72 + ka * 32, 72, l16, half, vf);
      o[t].v = wmma_bf16(pf, vf, o[t].v);
    }
  }

  // epilogue: window -> pixel rows, bf16 attn buffer [ROWS][256]
  const int nh_i = win >> 4, nw_i = win & 15;
  #pragma unroll
  for (int t = 0; t < 2; ++t) {
    int col = head * HD + t * 16 + l16;
    #pragma unroll
    for (int i = 0; i < 8; ++i) {
      int pos = wv * 16 + i + half * 8;
      int h = nh_i * 8 + (pos >> 3);
      int w = nw_i * 8 + (pos & 7);
      size_t row = ((size_t)bl * HPIX + h) * HPIX + w;
      attn_ws[row * DIMC + col] = f2bf(o[t].f[i]);
    }
  }
}

// ================= kernel 3: out = attn @ W_out + b_out (f32 out) =================
// grid (DIMC/64, ROWS/64), block 256
__global__ __launch_bounds__(256) void out_proj_kernel(
    const unsigned short* __restrict__ attn_ws, const float* __restrict__ Wout,
    const float* __restrict__ bout, float* __restrict__ out) {
  __shared__ unsigned short As[64 * 40];
  __shared__ unsigned short Bs[64 * 40];
  const int tid  = threadIdx.x;
  const int lane = tid & 31, l16 = lane & 15, half = lane >> 4;
  const int wid  = tid >> 5, wy = wid >> 1, wx = wid & 1;
  const int rowBase = blockIdx.y * 64;
  const int colBase = blockIdx.x * 64;

  FragC acc[2];
  #pragma unroll
  for (int t = 0; t < 2; ++t)
    #pragma unroll
    for (int i = 0; i < 8; ++i) acc[t].f[i] = 0.f;

  for (int k0 = 0; k0 < DIMC; k0 += 32) {
    // A: bf16 64x32 tile, async 16B chunk per thread into padded LDS
    {
      int r = tid >> 2, seg = tid & 3;
      const unsigned short* ap = attn_ws + (size_t)(rowBase + r) * DIMC + k0 + seg * 8;
      if (k0 + 32 < DIMC) __builtin_prefetch(ap + 32, 0, 1);
      async_copy16(ap, &As[r * 40 + seg * 8]);
    }
    // B: W_out f32 32x64 -> bf16 LDS transposed (N-major)
    #pragma unroll
    for (int rep = 0; rep < 2; ++rep) {
      int fidx = tid + rep * 256;
      int k = fidx >> 4, n4 = fidx & 15;
      float4 vw = *(const float4*)(Wout + (size_t)(k0 + k) * DIMC + colBase + n4 * 4);
      Bs[(n4 * 4 + 0) * 40 + k] = f2bf(vw.x);
      Bs[(n4 * 4 + 1) * 40 + k] = f2bf(vw.y);
      Bs[(n4 * 4 + 2) * 40 + k] = f2bf(vw.z);
      Bs[(n4 * 4 + 3) * 40 + k] = f2bf(vw.w);
    }
    wait_async0();
    __syncthreads();
    FragAB a, b0, b1;
    load_frag_a(As + wy * 16 * 40, 40, l16, half, a);
    load_frag_b(Bs + (wx * 32) * 40, 40, l16, half, b0);
    load_frag_b(Bs + (wx * 32 + 16) * 40, 40, l16, half, b1);
    acc[0].v = wmma_bf16(a, b0, acc[0].v);
    acc[1].v = wmma_bf16(a, b1, acc[1].v);
    __syncthreads();
  }

  #pragma unroll
  for (int t = 0; t < 2; ++t) {
    int cg = colBase + wx * 32 + t * 16 + l16;
    float bv = bout[cg];
    #pragma unroll
    for (int i = 0; i < 8; ++i) {
      int rg = rowBase + wy * 16 + i + half * 8;
      out[(size_t)rg * DIMC + cg] = acc[t].f[i] + bv;
    }
  }
}

// ================= host launch =================
extern "C" void kernel_launch(void* const* d_in, const int* in_sizes, int n_in,
                              void* d_out, int out_size, void* d_ws, size_t ws_size,
                              hipStream_t stream) {
  const float* x    = (const float*)d_in[0];
  const float* Wqkv = (const float*)d_in[1];
  const float* Wout = (const float*)d_in[2];
  const float* bout = (const float*)d_in[3];
  const float* pemb = (const float*)d_in[4];
  float* out = (float*)d_out;

  // workspace layout (bf16 elements): [q|k|v windowed] (3*SEL_STRIDE) + attn [ROWS*256]
  unsigned short* qkv_ws  = (unsigned short*)d_ws;
  unsigned short* attn_ws = qkv_ws + 3 * SEL_STRIDE;

  qkv_proj_kernel<<<dim3(NQKV / 64, ROWS / 64), 256, 0, stream>>>(x, Wqkv, qkv_ws);
  window_attn_kernel<<<dim3(NWIN, HEADS, BLD), 128, 0, stream>>>(qkv_ws, pemb, attn_ws);
  out_proj_kernel<<<dim3(DIMC / 64, ROWS / 64), 256, 0, stream>>>(attn_ws, Wout, bout, out);
}
